// CrystalGraphConvNet_38749194945199
// MI455X (gfx1250) — matrix operand order repaired and checked by reading, not compiled, not run on hardware
//
#include <hip/hip_runtime.h>
#include <hip/hip_bf16.h>

#define N_ATOMS 40000
#define E_EDGES 400000
#define N_CRYS  400
#define ORIG    93
#define NBRF    42
#define FDIM    128
#define NCONV   3
#define EPSV    1e-5f

#define KPAD     320   // padded K: 256 gathered + 48 edge (41 real) + 16 zero
#define EPADC    48    // padded edge-feature columns
#define AT_STRIDE 328  // LDS A-tile row stride (bf16 elems), +8 anti-conflict pad
#define TILE_M   64
#define KSTEPS   10

typedef __attribute__((ext_vector_type(16))) __bf16 v16bf;
typedef __attribute__((ext_vector_type(8)))  float  v8f;

union BF16x16 { uint4 q[2]; v16bf v; };

// ---- stats scratch layout (floats) ----
#define ST_SWA  0      // sum of atom weights (1)
#define ST_SWE  1      // sum of edge weights (1)
#define ST_S1   2      // edge-BN sum w*g      (256)
#define ST_S2   258    // edge-BN sum w*g^2    (256)
#define ST_A1   514    // folded edge-BN scale (256)
#define ST_BF   770    // folded edge-BN shift (256)
#define ST_S1A  1026   // atom-BN sum w*v      (128)
#define ST_S2A  1154   // atom-BN sum w*v^2    (128)
#define ST_A2   1282   // folded atom-BN scale (128)
#define ST_B2F  1410   // folded atom-BN shift (128)
#define ST_TOTAL 1538

__device__ __forceinline__ unsigned short f2bf(float f) {
  unsigned int u = __builtin_bit_cast(unsigned int, f);
  u += 0x7fffu + ((u >> 16) & 1u);
  return (unsigned short)(u >> 16);
}
__device__ __forceinline__ float softplusf(float x) {
  return log1pf(__expf(-fabsf(x))) + fmaxf(x, 0.0f);
}
__device__ __forceinline__ float sigmoidf(float x) {
  return 1.0f / (1.0f + __expf(-x));
}

// ---------------- small utility kernels ----------------
__global__ void k_zero(float* __restrict__ p, int n) {
  int i = blockIdx.x * 256 + threadIdx.x;
  if (i < n) p[i] = 0.0f;
}

__global__ void k_zero_layer_stats(float* __restrict__ stats) {
  int t = threadIdx.x;                 // 256 threads
  stats[ST_S1 + t] = 0.0f;
  stats[ST_S2 + t] = 0.0f;
  if (t < 128) { stats[ST_S1A + t] = 0.0f; stats[ST_S2A + t] = 0.0f; }
}

// sum of src[i*stride + offset] for i in [0,n) -> atomicAdd into *out
__global__ void k_reduce_strided(const float* __restrict__ src, int n, int stride,
                                 int offset, float* __restrict__ out) {
  __shared__ float L[256];
  float s = 0.0f;
  for (int i = blockIdx.x * 256 + threadIdx.x; i < n; i += gridDim.x * 256)
    s += src[(size_t)i * stride + offset];
  L[threadIdx.x] = s;
  __syncthreads();
  for (int w = 128; w > 0; w >>= 1) {
    if (threadIdx.x < w) L[threadIdx.x] += L[threadIdx.x + w];
    __syncthreads();
  }
  if (threadIdx.x == 0) atomicAdd(out, L[0]);
}

// ---------------- embedding: x = atom_fea[:,1:] @ emb_W + emb_b ----------------
__global__ __launch_bounds__(256) void k_embed(const float* __restrict__ atom_fea,
                                               const float* __restrict__ embW,
                                               const float* __restrict__ embB,
                                               float* __restrict__ x,
                                               unsigned short* __restrict__ xbf,
                                               float* __restrict__ w_atom) {
  int gid = blockIdx.x * 256 + threadIdx.x;   // exactly N_ATOMS*FDIM threads
  int a = gid >> 7, n = gid & 127;
  const float* row = atom_fea + (size_t)a * ORIG;
  float s = embB[n];
#pragma unroll 4
  for (int k = 0; k < ORIG - 1; ++k) s = fmaf(row[1 + k], embW[k * FDIM + n], s);
  x[gid] = s;
  xbf[gid] = f2bf(s);
  if (n == 0) w_atom[a] = row[0];
}

// ---------------- edge feature prep: bf16, padded 41 -> 48 cols ----------------
__global__ void k_prep_edges(const float* __restrict__ nbr_fea,
                             unsigned short* __restrict__ ebf) {
  int gid = blockIdx.x * 256 + threadIdx.x;   // exactly E*48 threads
  int e = gid / EPADC, c = gid - e * EPADC;
  float v = (c < NBRF - 1) ? nbr_fea[(size_t)e * NBRF + c] : 0.0f;
  ebf[gid] = f2bf(v);
}

// ------- pre-swizzle conv weights into WMMA B-fragment layout (bf16) -------
// layout: [layer][kt(10)][nt(16)][lane(32)][8 x u32]; per lane: n = nt*16+(lane&15),
// elem (2v+j) holds W[kt*32 + (lane>>4)*16 + 2v + j][n]  (zeros beyond K=297)
__global__ void k_prep_wfrag(const float* __restrict__ convW,
                             unsigned int* __restrict__ wfrag) {
  int b = blockIdx.x;                  // 3*10*16 = 480 blocks
  int l = b / 160, rem = b - l * 160;
  int kt = rem >> 4, nt = rem & 15;
  int lane = threadIdx.x >> 3, v = threadIdx.x & 7;
  int khalf = lane >> 4;
  int n = nt * 16 + (lane & 15);
  int k0 = kt * 32 + khalf * 16 + 2 * v;
  int k1 = k0 + 1;
  float w0 = (k0 < 297) ? convW[((size_t)l * 297 + k0) * 256 + n] : 0.0f;
  float w1 = (k1 < 297) ? convW[((size_t)l * 297 + k1) * 256 + n] : 0.0f;
  unsigned int packed = (unsigned int)f2bf(w0) | ((unsigned int)f2bf(w1) << 16);
  wfrag[((size_t)b * 32 + lane) * 8 + v] = packed;
}

// ------------- shared helpers for the two conv GEMM passes -------------
__device__ __forceinline__ void stage_tile(unsigned short* __restrict__ At,
                                           const unsigned short* __restrict__ xbf,
                                           const unsigned short* __restrict__ ebf,
                                           const int* __restrict__ nbr_idx,
                                           int e0, int tid) {
  // zero the K=304..319 pad (1024 elems)
  for (int i = tid; i < 1024; i += 256) {
    int r = i >> 4, c = 304 + (i & 15);
    At[r * AT_STRIDE + c] = 0;
  }
  // gather x rows: 64 edges * 2 atoms * 16 uint4 (128 bf16) = 2048 uint4
  for (int t = tid; t < 2048; t += 256) {
    int r = t >> 5;
    int h = (t >> 4) & 1;
    int chunk = t & 15;
    int av = nbr_idx[2 * (e0 + r) + h];
    const uint4* src = reinterpret_cast<const uint4*>(xbf + (size_t)av * FDIM + chunk * 8);
    uint4* dst = reinterpret_cast<uint4*>(At + r * AT_STRIDE + h * 128 + chunk * 8);
    *dst = *src;
  }
  // edge features: 64 edges * 6 uint4 (48 bf16) = 384 uint4
  for (int t = tid; t < 384; t += 256) {
    int r = t / 6, c = t - r * 6;
    const uint4* src = reinterpret_cast<const uint4*>(ebf + (size_t)(e0 + r) * EPADC + c * 8);
    uint4* dst = reinterpret_cast<uint4*>(At + r * AT_STRIDE + 256 + c * 8);
    *dst = *src;
  }
}

__device__ __forceinline__ void gemm_tile(const unsigned short* __restrict__ At,
                                          const unsigned int* __restrict__ wfragL,
                                          int wm, int wn, int lane, v8f acc[2][4]) {
  const int khalf = lane >> 4;
  const int lrow = lane & 15;
  for (int kt = 0; kt < KSTEPS; ++kt) {
    BF16x16 af[2];
#pragma unroll
    for (int mti = 0; mti < 2; ++mti) {
      int row = wm * 32 + mti * 16 + lrow;
      const uint4* pa =
          reinterpret_cast<const uint4*>(At + row * AT_STRIDE + kt * 32 + khalf * 16);
      af[mti].q[0] = pa[0];   // K: khalf*16 .. +7
      af[mti].q[1] = pa[1];   // K: khalf*16+8 .. +15
    }
    BF16x16 bf[4];
#pragma unroll
    for (int nti = 0; nti < 4; ++nti) {
      const uint4* pb = reinterpret_cast<const uint4*>(
          wfragL + ((size_t)((kt * 16 + wn * 4 + nti) * 32 + lane)) * 8);
      bf[nti].q[0] = pb[0];
      bf[nti].q[1] = pb[1];
    }
#pragma unroll
    for (int mti = 0; mti < 2; ++mti)
#pragma unroll
      for (int nti = 0; nti < 4; ++nti)
        acc[mti][nti] = __builtin_amdgcn_wmma_f32_16x16x32_bf16(
            false, af[mti].v, false, bf[nti].v, (short)0, acc[mti][nti], false, false);
  }
}

// ---- conv pass 1: GEMM + weighted sums for edge BN ----
__global__ __launch_bounds__(256) void k_conv_pass1(
    const unsigned short* __restrict__ xbf, const unsigned short* __restrict__ ebf,
    const int* __restrict__ nbr_idx, const float* __restrict__ nbr_fea,
    const unsigned int* __restrict__ wfragL, const float* __restrict__ convbL,
    float* __restrict__ stats) {
  __shared__ unsigned short At[TILE_M * AT_STRIDE];
  __shared__ float ew[TILE_M];
  __shared__ float S1loc[256], S2loc[256];
  int tid = threadIdx.x;
  int e0 = blockIdx.x * TILE_M;
  if (tid < TILE_M) ew[tid] = nbr_fea[(size_t)(e0 + tid) * NBRF + (NBRF - 1)];
  S1loc[tid] = 0.0f;
  S2loc[tid] = 0.0f;
  stage_tile(At, xbf, ebf, nbr_idx, e0, tid);
  __syncthreads();

  int lane = tid & 31, wid = tid >> 5;
  int wm = wid & 1, wn = wid >> 1;
  v8f acc[2][4] = {};
  gemm_tile(At, wfragL, wm, wn, lane, acc);

  int khalf = lane >> 4, lrow = lane & 15;
#pragma unroll
  for (int nti = 0; nti < 4; ++nti) {
    int n = wn * 64 + nti * 16 + lrow;
    float cb = convbL[n];
    float s1 = 0.0f, s2 = 0.0f;
#pragma unroll
    for (int mti = 0; mti < 2; ++mti)
#pragma unroll
      for (int r = 0; r < 8; ++r) {
        int row = wm * 32 + mti * 16 + khalf * 8 + r;
        float g = acc[mti][nti][r] + cb;
        float w = ew[row];
        s1 = fmaf(w, g, s1);
        s2 = fmaf(w * g, g, s2);
      }
    atomicAdd(&S1loc[n], s1);
    atomicAdd(&S2loc[n], s2);
  }
  __syncthreads();
  atomicAdd(&stats[ST_S1 + tid], S1loc[tid]);
  atomicAdd(&stats[ST_S2 + tid], S2loc[tid]);
}

// fold edge BN: a1 = g1*rsqrt(var+eps); bf = b1 + a1*(conv_b - mean)
__global__ void k_fin_edge(float* __restrict__ stats, const float* __restrict__ g1,
                           const float* __restrict__ b1, const float* __restrict__ cb) {
  int n = threadIdx.x;  // 256
  float swe = stats[ST_SWE];
  float mean = stats[ST_S1 + n] / swe;
  float var = stats[ST_S2 + n] / swe - mean * mean;
  float a1 = g1[n] * rsqrtf(var + EPSV);
  stats[ST_A1 + n] = a1;
  stats[ST_BF + n] = b1[n] + a1 * (cb[n] - mean);
}

// ---- conv pass 2: GEMM recompute + BN + gate + scatter-add ----
__global__ __launch_bounds__(256) void k_conv_pass2(
    const unsigned short* __restrict__ xbf, const unsigned short* __restrict__ ebf,
    const int* __restrict__ nbr_idx, const unsigned int* __restrict__ wfragL,
    const float* __restrict__ stats, float* __restrict__ summed) {
  __shared__ unsigned char smem[TILE_M * 256 * 4];  // 64 KB: A-tile, then g-buffer
  unsigned short* At = reinterpret_cast<unsigned short*>(smem);
  float* gbuf = reinterpret_cast<float*>(smem);
  int tid = threadIdx.x;
  int e0 = blockIdx.x * TILE_M;
  stage_tile(At, xbf, ebf, nbr_idx, e0, tid);
  __syncthreads();

  int lane = tid & 31, wid = tid >> 5;
  int wm = wid & 1, wn = wid >> 1;
  v8f acc[2][4] = {};
  gemm_tile(At, wfragL, wm, wn, lane, acc);
  __syncthreads();  // all waves finished reading A-tile; LDS becomes g-buffer

  int khalf = lane >> 4, lrow = lane & 15;
#pragma unroll
  for (int nti = 0; nti < 4; ++nti) {
    int n = wn * 64 + nti * 16 + lrow;
    float a1 = stats[ST_A1 + n];
    float bfold = stats[ST_BF + n];
#pragma unroll
    for (int mti = 0; mti < 2; ++mti)
#pragma unroll
      for (int r = 0; r < 8; ++r) {
        int row = wm * 32 + mti * 16 + khalf * 8 + r;
        gbuf[row * 256 + n] = fmaf(a1, acc[mti][nti][r], bfold);
      }
  }
  __syncthreads();

  // msg = sigmoid(filt)*softplus(core); scatter into summed[idx0]
  int m = tid >> 2, sub = tid & 3;
  int i0 = nbr_idx[2 * (e0 + m)];
  float* dst = summed + (size_t)i0 * FDIM;
#pragma unroll 4
  for (int j = 0; j < 32; ++j) {
    int n = sub * 32 + j;
    float f = gbuf[m * 256 + n];
    float c = gbuf[m * 256 + 128 + n];
    atomicAdd(dst + n, sigmoidf(f) * softplusf(c));
  }
}

// ---- atom BN stats over summed ----
__global__ __launch_bounds__(256) void k_atom_stats(const float* __restrict__ summed,
                                                    const float* __restrict__ w_atom,
                                                    float* __restrict__ stats) {
  __shared__ float L1[128], L2[128];
  int tid = threadIdx.x;
  if (tid < 128) { L1[tid] = 0.0f; L2[tid] = 0.0f; }
  __syncthreads();
  int n = tid & 127, half = tid >> 7;
  float s1 = 0.0f, s2 = 0.0f;
  for (int a = blockIdx.x * 2 + half; a < N_ATOMS; a += gridDim.x * 2) {
    float w = w_atom[a];
    float v = summed[(size_t)a * FDIM + n];
    s1 = fmaf(w, v, s1);
    s2 = fmaf(w * v, v, s2);
  }
  atomicAdd(&L1[n], s1);
  atomicAdd(&L2[n], s2);
  __syncthreads();
  if (tid < 128) {
    atomicAdd(&stats[ST_S1A + tid], L1[tid]);
    atomicAdd(&stats[ST_S2A + tid], L2[tid]);
  }
}

__global__ void k_fin_atom(float* __restrict__ stats, const float* __restrict__ g2,
                           const float* __restrict__ b2) {
  int n = threadIdx.x;  // 128
  float swa = stats[ST_SWA];
  float mean = stats[ST_S1A + n] / swa;
  float var = stats[ST_S2A + n] / swa - mean * mean;
  float a2 = g2[n] * rsqrtf(var + EPSV);
  stats[ST_A2 + n] = a2;
  stats[ST_B2F + n] = b2[n] - a2 * mean;
}

// x = softplus(x + a2*summed + b2f); refresh x_bf16
__global__ __launch_bounds__(256) void k_update_x(float* __restrict__ x,
                                                  const float* __restrict__ summed,
                                                  const float* __restrict__ stats,
                                                  unsigned short* __restrict__ xbf) {
  int gid = blockIdx.x * 256 + threadIdx.x;
  int n = gid & 127;
  float v = x[gid] + fmaf(stats[ST_A2 + n], summed[gid], stats[ST_B2F + n]);
  float r = softplusf(v);
  x[gid] = r;
  xbf[gid] = f2bf(r);
}

// ---- pooling (crystals are contiguous blocks of 100 atoms) + MLP head ----
__global__ void k_pool(const float* __restrict__ x, const float* __restrict__ w_atom,
                       float* __restrict__ spc) {
  int c = blockIdx.x, n = threadIdx.x;  // 128 threads
  float acc = 0.0f, sw = 0.0f;
  for (int a = 0; a < N_ATOMS / N_CRYS; ++a) {
    float w = w_atom[c * (N_ATOMS / N_CRYS) + a];
    sw += w;
    acc = fmaf(w, x[(size_t)(c * (N_ATOMS / N_CRYS) + a) * FDIM + n], acc);
  }
  spc[c * FDIM + n] = softplusf(acc / sw);
}

__global__ __launch_bounds__(256) void k_fc1(const float* __restrict__ spc,
                                             const float* __restrict__ W,
                                             const float* __restrict__ b,
                                             float* __restrict__ h1) {
  __shared__ float sp[128];
  int c = blockIdx.x, t = threadIdx.x;
  if (t < 128) sp[t] = spc[c * 128 + t];
  __syncthreads();
  float s = b[t];
#pragma unroll 4
  for (int k = 0; k < 128; ++k) s = fmaf(sp[k], W[k * 256 + t], s);
  h1[c * 256 + t] = softplusf(s);
}

__global__ __launch_bounds__(256) void k_fc2(const float* __restrict__ h1,
                                             const float* __restrict__ W,
                                             const float* __restrict__ b,
                                             float* __restrict__ h2) {
  __shared__ float hh[256];
  int c = blockIdx.x, t = threadIdx.x;
  hh[t] = h1[c * 256 + t];
  __syncthreads();
  float s = b[t];
#pragma unroll 4
  for (int k = 0; k < 256; ++k) s = fmaf(hh[k], W[k * 256 + t], s);
  h2[c * 256 + t] = softplusf(s);
}

__global__ void k_out(const float* __restrict__ h2, const float* __restrict__ Wout,
                      const float* __restrict__ bout, float* __restrict__ out) {
  __shared__ float L[256];
  int c = blockIdx.x, t = threadIdx.x;
  L[t] = h2[c * 256 + t] * Wout[t];
  __syncthreads();
  for (int w = 128; w > 0; w >>= 1) {
    if (t < w) L[t] += L[t + w];
    __syncthreads();
  }
  if (t == 0) out[c] = L[0] + bout[0];
}

// ---------------------------------------------------------------------------
extern "C" void kernel_launch(void* const* d_in, const int* in_sizes, int n_in,
                              void* d_out, int out_size, void* d_ws, size_t ws_size,
                              hipStream_t stream) {
  (void)in_sizes; (void)n_in; (void)out_size; (void)ws_size;
  const float* atom_fea = (const float*)d_in[0];
  const float* nbr_fea  = (const float*)d_in[1];
  const int*   nbr_idx  = (const int*)d_in[2];
  const float* emb_W    = (const float*)d_in[4];
  const float* emb_b    = (const float*)d_in[5];
  const float* conv_W   = (const float*)d_in[6];
  const float* conv_b   = (const float*)d_in[7];
  const float* bn1_g    = (const float*)d_in[8];
  const float* bn1_b    = (const float*)d_in[9];
  const float* bn2_g    = (const float*)d_in[10];
  const float* bn2_b    = (const float*)d_in[11];
  const float* fc1_W    = (const float*)d_in[12];
  const float* fc1_b    = (const float*)d_in[13];
  const float* fc2_W    = (const float*)d_in[14];
  const float* fc2_b    = (const float*)d_in[15];
  const float* out_W    = (const float*)d_in[16];
  const float* out_b    = (const float*)d_in[17];
  float* out = (float*)d_out;

  // workspace carve-up (256B aligned)
  char* ws = (char*)d_ws;
  size_t off = 0;
  auto alloc = [&](size_t bytes) {
    void* p = ws + off;
    off += (bytes + 255) & ~(size_t)255;
    return p;
  };
  const size_t XN = (size_t)N_ATOMS * FDIM;         // 5,120,000
  float*          x      = (float*)alloc(XN * 4);
  float*          summed = (float*)alloc(XN * 4);
  float*          w_atom = (float*)alloc((size_t)N_ATOMS * 4);
  float*          stats  = (float*)alloc(ST_TOTAL * 4);
  float*          spc    = (float*)alloc((size_t)N_CRYS * FDIM * 4);
  float*          h1     = (float*)alloc((size_t)N_CRYS * 256 * 4);
  float*          h2     = (float*)alloc((size_t)N_CRYS * 256 * 4);
  unsigned short* xbf    = (unsigned short*)alloc(XN * 2);
  unsigned short* ebf    = (unsigned short*)alloc((size_t)E_EDGES * EPADC * 2);
  unsigned int*   wfrag  = (unsigned int*)alloc((size_t)NCONV * 10 * 16 * 32 * 8 * 4);

  const int xblocks = (int)(XN / 256);                         // 20000
  const int etiles  = E_EDGES / TILE_M;                        // 6250

  k_zero<<<(ST_TOTAL + 255) / 256, 256, 0, stream>>>(stats, ST_TOTAL);
  k_reduce_strided<<<256, 256, 0, stream>>>(atom_fea, N_ATOMS, ORIG, 0, stats + ST_SWA);
  k_reduce_strided<<<256, 256, 0, stream>>>(nbr_fea, E_EDGES, NBRF, NBRF - 1, stats + ST_SWE);
  k_embed<<<xblocks, 256, 0, stream>>>(atom_fea, emb_W, emb_b, x, xbf, w_atom);
  k_prep_edges<<<(E_EDGES * EPADC) / 256, 256, 0, stream>>>(nbr_fea, ebf);
  k_prep_wfrag<<<NCONV * 10 * 16, 256, 0, stream>>>(conv_W, wfrag);

  for (int l = 0; l < NCONV; ++l) {
    const unsigned int* wfragL = wfrag + (size_t)l * 10 * 16 * 32 * 8;
    k_zero_layer_stats<<<1, 256, 0, stream>>>(stats);
    k_conv_pass1<<<etiles, 256, 0, stream>>>(xbf, ebf, nbr_idx, nbr_fea, wfragL,
                                             conv_b + l * 256, stats);
    k_fin_edge<<<1, 256, 0, stream>>>(stats, bn1_g + l * 256, bn1_b + l * 256,
                                      conv_b + l * 256);
    k_zero<<<xblocks, 256, 0, stream>>>(summed, (int)XN);
    k_conv_pass2<<<etiles, 256, 0, stream>>>(xbf, ebf, nbr_idx, wfragL, stats, summed);
    k_atom_stats<<<512, 256, 0, stream>>>(summed, w_atom, stats);
    k_fin_atom<<<1, 128, 0, stream>>>(stats, bn2_g + l * 128, bn2_b + l * 128);
    k_update_x<<<xblocks, 256, 0, stream>>>(x, summed, stats, xbf);
  }

  k_pool<<<N_CRYS, 128, 0, stream>>>(x, w_atom, spc);
  k_fc1<<<N_CRYS, 256, 0, stream>>>(spc, fc1_W, fc1_b, h1);
  k_fc2<<<N_CRYS, 256, 0, stream>>>(h1, fc2_W, fc2_b, h2);
  k_out<<<N_CRYS, 256, 0, stream>>>(h2, out_W, out_b, out);
}
